// Attention_64098091925576
// MI455X (gfx1250) — matrix operand order repaired and checked by reading, not compiled, not run on hardware
//
#include <hip/hip_runtime.h>
#include <hip/hip_bf16.h>
#include <math.h>

typedef _Float16 half_t;
typedef __attribute__((ext_vector_type(16))) _Float16 v16h;
typedef __attribute__((ext_vector_type(8)))  _Float16 v8h;
typedef __attribute__((ext_vector_type(8)))  float    v8f;

#define B_  2
#define S_  2048
#define D_  4096
#define H_  32
#define KVH_ 8
#define HD_ 128
#define NREP_ (H_ / KVH_)

// ---------------------------------------------------------------------------
// helpers
// ---------------------------------------------------------------------------
__device__ __forceinline__ v16h cat8(v8h lo, v8h hi) {
  v16h r;
#pragma unroll
  for (int i = 0; i < 8; ++i) { r[i] = lo[i]; r[i + 8] = hi[i]; }
  return r;
}

// A-matrix fragment (16x32 f16): lane m = lane%16, hi = lane/16.
// elements 0..7  = K = hi*8 + 0..7 ; elements 8..15 = K = 16 + hi*8 + 0..7
__device__ __forceinline__ v16h load_a_frag(const half_t* rowptr, int hi) {
  const half_t* p = rowptr + hi * 8;
  v8h lo = *(const v8h*)p;
  v8h hh = *(const v8h*)(p + 16);
  return cat8(lo, hh);
}

// B-matrix fragment (32x16 f16): lane n = lane%16, hi = lane/16.
// elements 0..7  = K = hi*8 + 0..7 ; elements 8..15 = K = 16 + hi*8 + 0..7
__device__ __forceinline__ v16h load_b_frag(const half_t* rowptr, int hi) {
  const half_t* p = rowptr + hi * 8;
  v8h lo = *(const v8h*)p;
  v8h hh = *(const v8h*)(p + 16);
  return cat8(lo, hh);
}

__device__ __forceinline__ v8f wmma_f16(v16h a, v16h b, v8f c) {
  return __builtin_amdgcn_wmma_f32_16x16x32_f16(false, a, false, b, (short)0, c,
                                                false, false);
}

// Async global->LDS 16-byte copy (gfx1250, tracked by ASYNCcnt).
// Low 32 bits of a generic LDS pointer are the LDS byte offset (aperture rule).
__device__ __forceinline__ void async_cp16(const half_t* g, half_t* l) {
  unsigned ldsoff = (unsigned)(unsigned long long)(uintptr_t)l;
  unsigned long long ga = (unsigned long long)(uintptr_t)g;
  asm volatile("global_load_async_to_lds_b128 %0, %1, off"
               :: "v"(ldsoff), "v"(ga) : "memory");
}
#define WAIT_ASYNCCNT(n) asm volatile("s_wait_asynccnt " n ::: "memory")

// ---------------------------------------------------------------------------
// f32 -> f16 conversion
// ---------------------------------------------------------------------------
__global__ void cvt_f32_f16(const float* __restrict__ in, half_t* __restrict__ out,
                            size_t n) {
  size_t i = (size_t)blockIdx.x * blockDim.x + threadIdx.x;
  size_t stride = (size_t)gridDim.x * blockDim.x;
  for (; i < n; i += stride) out[i] = (half_t)in[i];
}

// ---------------------------------------------------------------------------
// RoPE + layout transform: raw [B*S, nh*HD] -> out [B, nh, S, HD] (f16)
// ---------------------------------------------------------------------------
__global__ void rope_perm(const half_t* __restrict__ raw,
                          const float* __restrict__ cosT,
                          const float* __restrict__ sinT,
                          half_t* __restrict__ out, int nh, float scale) {
  size_t idx = (size_t)blockIdx.x * blockDim.x + threadIdx.x;
  size_t total = (size_t)B_ * S_ * nh * (HD_ / 2);
  if (idx >= total) return;
  int p = idx % (HD_ / 2);
  size_t t = idx / (HD_ / 2);
  int hh = t % nh;
  size_t bs = t / nh;
  int s = bs % S_;
  int b = bs / S_;
  const half_t* rp = raw + bs * ((size_t)nh * HD_) + (size_t)hh * HD_ + 2 * p;
  float t0 = (float)rp[0], t1 = (float)rp[1];
  float c = cosT[(size_t)s * (HD_ / 2) + p];
  float sn = sinT[(size_t)s * (HD_ / 2) + p];
  float o0 = (t0 * c - t1 * sn) * scale;
  float o1 = (t0 * sn + t1 * c) * scale;
  half_t* op = out + (((size_t)(b * nh + hh) * S_ + s) * HD_) + 2 * p;
  op[0] = (half_t)o0;
  op[1] = (half_t)o1;
}

// V transpose: raw [B*S, KVH*HD] -> out [B, KVH, HD, S] (f16)
__global__ void v_trans(const half_t* __restrict__ raw, half_t* __restrict__ out) {
  size_t idx = (size_t)blockIdx.x * blockDim.x + threadIdx.x;
  size_t total = (size_t)B_ * S_ * KVH_ * HD_;
  if (idx >= total) return;
  int d = idx % HD_;
  size_t t = idx / HD_;
  int kv = t % KVH_;
  size_t bs = t / KVH_;
  int s = bs % S_;
  int b = bs / S_;
  out[((size_t)(b * KVH_ + kv) * HD_ + d) * S_ + s] =
      raw[bs * ((size_t)KVH_ * HD_) + (size_t)kv * HD_ + d];
}

// ---------------------------------------------------------------------------
// Tiled WMMA GEMM: C[M,N] = A[M,K] * Bw[N,K]^T   (A,Bw f16, K-contiguous)
// Block 256 thr = 8 waves (2x4), tile 128x128x32; wave tile 64x32 (4x2 frags).
// Double-buffered via GLOBAL_LOAD_ASYNC_TO_LDS_B128 (ASYNCcnt).
// ---------------------------------------------------------------------------
#define BK_ 32
#define LDT_ (BK_ + 8)  // padded LDS row stride (halfs)

template <bool OUT_F32>
__global__ __launch_bounds__(256) void gemm_wmma(const half_t* __restrict__ A,
                                                 const half_t* __restrict__ Bw,
                                                 void* __restrict__ Cout, int M,
                                                 int N, int K) {
  __shared__ half_t As[2][128 * LDT_];
  __shared__ half_t Bs[2][128 * LDT_];
  const int tid = threadIdx.x;
  const int lane = tid & 31, wid = tid >> 5;
  const int wm = wid >> 2, wn = wid & 3;  // 2 x 4 wave grid
  const int lm = lane & 15, hi = lane >> 4;
  const int rowblk = blockIdx.y * 128, colblk = blockIdx.x * 128;

  const v8f vzero = {0.f, 0.f, 0.f, 0.f, 0.f, 0.f, 0.f, 0.f};
  v8f acc[4][2];
#pragma unroll
  for (int i = 0; i < 4; ++i)
#pragma unroll
    for (int j = 0; j < 2; ++j) acc[i][j] = vzero;

  const int lr = tid >> 1;        // 0..127
  const int lc = (tid & 1) * 16;  // 0 or 16
  const half_t* gA = A + (size_t)(rowblk + lr) * K + lc;
  const half_t* gB = Bw + (size_t)(colblk + lr) * K + lc;

  // prologue: stage tile 0 into buffer 0 (4 async b128 copies per thread)
  {
    half_t* lA = &As[0][lr * LDT_ + lc];
    half_t* lB = &Bs[0][lr * LDT_ + lc];
    async_cp16(gA, lA);
    async_cp16(gA + 8, lA + 8);
    async_cp16(gB, lB);
    async_cp16(gB + 8, lB + 8);
  }

  const int T = K / BK_;
  for (int kt = 0; kt < T; ++kt) {
    const int cur = kt & 1;
    if (kt + 1 < T) {  // stage next tile into the other buffer
      const int nk = (kt + 1) * BK_;
      half_t* nA = &As[cur ^ 1][lr * LDT_ + lc];
      half_t* nB = &Bs[cur ^ 1][lr * LDT_ + lc];
      async_cp16(gA + nk, nA);
      async_cp16(gA + nk + 8, nA + 8);
      async_cp16(gB + nk, nB);
      async_cp16(gB + nk + 8, nB + 8);
      WAIT_ASYNCCNT("4");  // tile kt's 4 copies done; next tile in flight
    } else {
      WAIT_ASYNCCNT("0");
    }
    __syncthreads();  // all waves' staged data visible

    v16h af[4], bf[2];
#pragma unroll
    for (int i = 0; i < 4; ++i)
      af[i] = load_a_frag(&As[cur][(wm * 64 + i * 16 + lm) * LDT_], hi);
#pragma unroll
    for (int j = 0; j < 2; ++j)
      bf[j] = load_b_frag(&Bs[cur][(wn * 32 + j * 16 + lm) * LDT_], hi);
#pragma unroll
    for (int i = 0; i < 4; ++i)
#pragma unroll
      for (int j = 0; j < 2; ++j) acc[i][j] = wmma_f16(af[i], bf[j], acc[i][j]);
    __syncthreads();  // done reading buf `cur` before it is restaged
  }

#pragma unroll
  for (int i = 0; i < 4; ++i)
#pragma unroll
    for (int j = 0; j < 2; ++j)
#pragma unroll
      for (int r = 0; r < 8; ++r) {
        int row = rowblk + wm * 64 + i * 16 + r + 8 * hi;
        int col = colblk + wn * 32 + j * 16 + lm;
        if (OUT_F32)
          ((float*)Cout)[(size_t)row * N + col] = acc[i][j][r];
        else
          ((half_t*)Cout)[(size_t)row * N + col] = (half_t)acc[i][j][r];
      }
}

// ---------------------------------------------------------------------------
// Flash attention: Q [B,H,S,HD] (pre-scaled), K [B,KVH,S,HD], Vt [B,KVH,HD,S]
// -> AO f16 [B*S, H*HD].  Each wave owns 16 queries; KV streamed 32 at a time.
// ---------------------------------------------------------------------------
#define LDP_ 40  // padded P-tile row stride (halfs)

__global__ __launch_bounds__(256) void attn_wmma(const half_t* __restrict__ Q,
                                                 const half_t* __restrict__ Kk,
                                                 const half_t* __restrict__ Vt,
                                                 half_t* __restrict__ AO) {
  __shared__ half_t Pst[8 * 16 * LDP_];
  const int lane = threadIdx.x & 31, wid = threadIdx.x >> 5;
  const int lm = lane & 15, hi = lane >> 4;
  const int qbase = blockIdx.x * 128 + wid * 16;
  const int h = blockIdx.y, b = blockIdx.z;
  const int kvh = h / NREP_;
  const int pbase = wid * 16 * LDP_;

  const half_t* Qp = Q + ((size_t)(b * H_ + h) * S_ + qbase) * HD_;
  const half_t* Kp = Kk + (size_t)(b * KVH_ + kvh) * S_ * HD_;
  const half_t* Vp = Vt + (size_t)(b * KVH_ + kvh) * HD_ * S_;

  // Q fragments: 16 x 128, kept in registers for the whole pass
  v16h qf[4];
#pragma unroll
  for (int c = 0; c < 4; ++c)
    qf[c] = load_a_frag(Qp + (size_t)lm * HD_ + c * 32, hi);

  const v8f vzero = {0.f, 0.f, 0.f, 0.f, 0.f, 0.f, 0.f, 0.f};
  v8f o[8];
#pragma unroll
  for (int ch = 0; ch < 8; ++ch) o[ch] = vzero;
  float mrow[8], lrow[8];
#pragma unroll
  for (int r = 0; r < 8; ++r) { mrow[r] = -1e30f; lrow[r] = 0.f; }

  const int qend = qbase + 16;  // causal bound (wave-uniform)
  for (int kv0 = 0; kv0 < qend; kv0 += 32) {
    // prefetch next KV tile while this one is consumed (global_prefetch_b8)
    if (kv0 + 32 < qend) {
      __builtin_prefetch(Kp + (size_t)(kv0 + 32 + lane) * HD_, 0, 3);
      __builtin_prefetch(Vp + (size_t)lane * S_ + kv0 + 32, 0, 3);
      __builtin_prefetch(Vp + (size_t)(64 + lane) * S_ + kv0 + 32, 0, 3);
    }

    // ---- scores: S(16x32) = Q(16x128) * K^T(128x32), two 16x16 C-tiles ----
    v8f s0 = vzero, s1 = vzero;
#pragma unroll
    for (int c = 0; c < 4; ++c) {
      v16h kf0 = load_b_frag(Kp + (size_t)(kv0 + lm) * HD_ + c * 32, hi);
      s0 = wmma_f16(qf[c], kf0, s0);
      v16h kf1 = load_b_frag(Kp + (size_t)(kv0 + 16 + lm) * HD_ + c * 32, hi);
      s1 = wmma_f16(qf[c], kf1, s1);
    }

    // ---- online softmax (rows split across half-waves; 16-lane reductions) --
#pragma unroll
    for (int r = 0; r < 8; ++r) {
      const int m = r + 8 * hi;
      const int q = qbase + m;
      float a0 = s0[r]; if (kv0 + lm > q)      a0 = -1e30f;
      float a1 = s1[r]; if (kv0 + 16 + lm > q) a1 = -1e30f;
      float mx = fmaxf(a0, a1);
#pragma unroll
      for (int off = 8; off; off >>= 1) mx = fmaxf(mx, __shfl_xor(mx, off, 32));
      const float mnew = fmaxf(mrow[r], mx);
      const float alpha = expf(mrow[r] - mnew);
      const float p0 = expf(a0 - mnew);
      const float p1 = expf(a1 - mnew);
      float ps = p0 + p1;
#pragma unroll
      for (int off = 8; off; off >>= 1) ps += __shfl_xor(ps, off, 32);
      lrow[r] = lrow[r] * alpha + ps;
      mrow[r] = mnew;
#pragma unroll
      for (int ch = 0; ch < 8; ++ch) o[ch][r] *= alpha;
      Pst[pbase + m * LDP_ + lm] = (half_t)p0;
      Pst[pbase + m * LDP_ + 16 + lm] = (half_t)p1;
    }
    asm volatile("" ::: "memory");  // same-wave LDS ops are HW in-order

    // ---- out update: O(16x128) += P(16x32) * V(32x128) ----
    v16h pf = load_a_frag(&Pst[pbase + lm * LDP_], hi);
#pragma unroll
    for (int ch = 0; ch < 8; ++ch) {
      v16h vf = load_b_frag(Vp + (size_t)(ch * 16 + lm) * S_ + kv0, hi);
      o[ch] = wmma_f16(pf, vf, o[ch]);
    }
    asm volatile("" ::: "memory");
  }

  // ---- finalize: divide by row sum, emit f16 to [B*S, H*HD] ----
#pragma unroll
  for (int r = 0; r < 8; ++r) {
    const int m = r + 8 * hi;
    const float inv = 1.0f / lrow[r];
    const size_t row = (size_t)(b * S_ + qbase + m) * (H_ * HD_) + h * HD_;
#pragma unroll
    for (int ch = 0; ch < 8; ++ch)
      AO[row + ch * 16 + lm] = (half_t)(o[ch][r] * inv);
  }
}

// ---------------------------------------------------------------------------
// host launch
// ---------------------------------------------------------------------------
extern "C" void kernel_launch(void* const* d_in, const int* in_sizes, int n_in,
                              void* d_out, int out_size, void* d_ws, size_t ws_size,
                              hipStream_t stream) {
  const float* x_norm = (const float*)d_in[0];
  const float* wq = (const float*)d_in[1];
  const float* wk = (const float*)d_in[2];
  const float* wv = (const float*)d_in[3];
  const float* wo = (const float*)d_in[4];
  const float* fcos = (const float*)d_in[5];
  const float* fsin = (const float*)d_in[6];
  // d_in[7] = mask (handled analytically), d_in[8] = start_pos (==0)

  half_t* ws = (half_t*)d_ws;
  const size_t NX = (size_t)B_ * S_ * D_;           // 16,777,216
  const size_t NKV = (size_t)B_ * S_ * KVH_ * HD_;  // 4,194,304
  half_t* X16  = ws;                 // [B*S, D]
  half_t* W16  = X16 + NX;           // reusable weight buffer (max D*D)
  half_t* QRAW = W16 + NX;           // [B*S, H*HD]  (reused later as AO)
  half_t* QR   = QRAW + NX;          // [B, H, S, HD]
  half_t* KRAW = QR + NX;            // [B*S, KVH*HD]
  half_t* KR   = KRAW + NKV;         // [B, KVH, S, HD]
  half_t* VRAW = KR + NKV;           // [B*S, KVH*HD]
  half_t* VT   = VRAW + NKV;         // [B, KVH, HD, S]
  half_t* AO   = QRAW;               // reuse

  const int M = B_ * S_;  // 4096
  dim3 blk(256);
  auto cvtgrid = [](size_t n) { return dim3((unsigned)((n + 255) / 256)); };

  // x -> f16
  cvt_f32_f16<<<cvtgrid(NX), blk, 0, stream>>>(x_norm, X16, NX);

  // Q = x * wq^T ; RoPE(+1/sqrt(HD)) ; -> [B,H,S,HD]
  cvt_f32_f16<<<cvtgrid((size_t)H_ * HD_ * D_), blk, 0, stream>>>(
      wq, W16, (size_t)H_ * HD_ * D_);
  gemm_wmma<false><<<dim3((H_ * HD_) / 128, M / 128), blk, 0, stream>>>(
      X16, W16, QRAW, M, H_ * HD_, D_);
  rope_perm<<<cvtgrid((size_t)B_ * S_ * H_ * (HD_ / 2)), blk, 0, stream>>>(
      QRAW, fcos, fsin, QR, H_, 1.0f / sqrtf((float)HD_));

  // K = x * wk^T ; RoPE ; -> [B,KVH,S,HD]
  cvt_f32_f16<<<cvtgrid((size_t)KVH_ * HD_ * D_), blk, 0, stream>>>(
      wk, W16, (size_t)KVH_ * HD_ * D_);
  gemm_wmma<false><<<dim3((KVH_ * HD_) / 128, M / 128), blk, 0, stream>>>(
      X16, W16, KRAW, M, KVH_ * HD_, D_);
  rope_perm<<<cvtgrid((size_t)B_ * S_ * KVH_ * (HD_ / 2)), blk, 0, stream>>>(
      KRAW, fcos, fsin, KR, KVH_, 1.0f);

  // V = x * wv^T ; transpose -> [B,KVH,HD,S]
  cvt_f32_f16<<<cvtgrid((size_t)KVH_ * HD_ * D_), blk, 0, stream>>>(
      wv, W16, (size_t)KVH_ * HD_ * D_);
  gemm_wmma<false><<<dim3((KVH_ * HD_) / 128, M / 128), blk, 0, stream>>>(
      X16, W16, VRAW, M, KVH_ * HD_, D_);
  v_trans<<<cvtgrid(NKV), blk, 0, stream>>>(VRAW, VT);

  // attention -> AO f16 [B*S, H*HD]
  attn_wmma<<<dim3(S_ / 128, H_, B_), blk, 0, stream>>>(QR, KR, VT, AO);

  // out = AO * wo^T -> f32 d_out [B,S,D]
  cvt_f32_f16<<<cvtgrid((size_t)D_ * H_ * HD_), blk, 0, stream>>>(
      wo, W16, (size_t)D_ * H_ * HD_);
  gemm_wmma<true><<<dim3(D_ / 128, M / 128), blk, 0, stream>>>(
      AO, W16, (float*)d_out, M, D_, H_ * HD_);
}
